// Head_72627896975586
// MI455X (gfx1250) — matrix operand order repaired and checked by reading, not compiled, not run on hardware
//
#include <hip/hip_runtime.h>
#include <math.h>

typedef __attribute__((ext_vector_type(16))) _Float16 v16h;
typedef __attribute__((ext_vector_type(8)))  float    v8f;
typedef int v4i __attribute__((vector_size(16)));

#define HD       64
#define NEG_BIG  (-1e30f)
#define KSTRIDE  68                 // padded LDS row stride (floats) for K/V tiles
#define CHUNKF   (32 * KSTRIDE)     // one 32-row chunk in LDS (floats)
#define PSTRIDE  34                 // padded stride for P-transpose staging

// ---- async global->LDS 16B copy (ASYNCcnt-tracked) ----
// builtin signature (from clang diagnostic): (v4i AS1*, v4i AS3*, imm, imm)
#define ASYNC_B128(g, l)                                                       \
  __builtin_amdgcn_global_load_async_to_lds_b128(                              \
      (__attribute__((address_space(1))) v4i*)(g),                             \
      (__attribute__((address_space(3))) v4i*)(l), 0, 0)

// ---- DPP16 reductions within each 16-lane half-wave ----
// xor1=quad_perm[1,0,3,2](0xB1), xor2=quad_perm[2,3,0,1](0x4E),
// xor7=row_half_mirror(0x141), xor15=row_mirror(0x140)
__device__ __forceinline__ float red_max16(float v) {
  int x;
  x = __builtin_bit_cast(int, v);
  v = fmaxf(v, __builtin_bit_cast(float, __builtin_amdgcn_update_dpp(x, x, 0x0B1, 0xF, 0xF, false)));
  x = __builtin_bit_cast(int, v);
  v = fmaxf(v, __builtin_bit_cast(float, __builtin_amdgcn_update_dpp(x, x, 0x04E, 0xF, 0xF, false)));
  x = __builtin_bit_cast(int, v);
  v = fmaxf(v, __builtin_bit_cast(float, __builtin_amdgcn_update_dpp(x, x, 0x141, 0xF, 0xF, false)));
  x = __builtin_bit_cast(int, v);
  v = fmaxf(v, __builtin_bit_cast(float, __builtin_amdgcn_update_dpp(x, x, 0x140, 0xF, 0xF, false)));
  return v;
}
__device__ __forceinline__ float red_sum16(float v) {
  int x;
  x = __builtin_bit_cast(int, v);
  v += __builtin_bit_cast(float, __builtin_amdgcn_update_dpp(x, x, 0x0B1, 0xF, 0xF, false));
  x = __builtin_bit_cast(int, v);
  v += __builtin_bit_cast(float, __builtin_amdgcn_update_dpp(x, x, 0x04E, 0xF, 0xF, false));
  x = __builtin_bit_cast(int, v);
  v += __builtin_bit_cast(float, __builtin_amdgcn_update_dpp(x, x, 0x141, 0xF, 0xF, false));
  x = __builtin_bit_cast(int, v);
  v += __builtin_bit_cast(float, __builtin_amdgcn_update_dpp(x, x, 0x140, 0xF, 0xF, false));
  return v;
}

// Block = 4 wave32 = 64 query rows of one batch. K/V 32-key chunks are
// double-buffered into LDS via GLOBAL_LOAD_ASYNC_TO_LDS_B128; each wave runs
// a 16-row flash-attention step (v_wmma_f32_16x16x32_f16) out of LDS.
__global__ __launch_bounds__(128)
void flash_attn_causal_wmma_async(const float* __restrict__ K,
                                  const float* __restrict__ Q,
                                  const float* __restrict__ V,
                                  float* __restrict__ O,
                                  int T, float scale) {
  __shared__ float kvbuf[2][2 * CHUNKF];     // [buf][ K tile | V tile ]
  __shared__ float ptile[4][16 * PSTRIDE];   // per-wave P transpose staging

  const int tid  = threadIdx.x;              // 0..127
  const int lane = tid & 31;
  const int wib  = tid >> 5;                 // wave in block
  const int col  = lane & 15;                // N-col (B/C/D) or A-row
  const int hi   = lane >> 4;

  const int qtiles64 = T >> 6;
  const int b        = blockIdx.x / qtiles64;
  const int q_base   = (blockIdx.x % qtiles64) << 6;  // block's 64 rows
  const int qw_base  = q_base + (wib << 4);           // this wave's 16 rows

  const float* Kb = K + (size_t)b * T * HD;
  const float* Qb = Q + (size_t)b * T * HD;
  const float* Vb = V + (size_t)b * T * HD;
  float*       Ob = O + (size_t)b * T * HD;

  // ---- stage one 32-key K+V chunk (2x8KB, contiguous in global) ----
  auto stage = [&](int ci, float* dst) {
#pragma unroll
    for (int f = 0; f < 4; ++f) {
      const int s   = f * 128 + tid;            // 16B slot (0..511)
      const int row = s >> 4, sir = s & 15;
      const int lo  = row * KSTRIDE + sir * 4;  // padded LDS float index
      ASYNC_B128(Kb + ((size_t)ci << 5) * HD + s * 4, dst + lo);
      ASYNC_B128(Vb + ((size_t)ci << 5) * HD + s * 4, dst + CHUNKF + lo);
    }
  };

  // ---- Q tile (16x64) -> two f16 A operands, pre-scaled by n_embed^-0.5 ----
  v16h aq0, aq1;
  {
    const float* qrow = Qb + (size_t)(qw_base + col) * HD;
#pragma unroll
    for (int e = 0; e < 16; ++e) {
      int j  = e >> 1, lo = e & 1;
      int kk = (hi << 4) + e;  // A operand, contiguous k = 16*hi + e
      aq0[e] = (_Float16)(qrow[kk] * scale);
      aq1[e] = (_Float16)(qrow[kk + 32] * scale);
    }
  }

  v8f oacc[4] = {v8f{}, v8f{}, v8f{}, v8f{}};  // 16x64 output accumulator
  float mrow[8], lrow[8];
#pragma unroll
  for (int j = 0; j < 8; ++j) { mrow[j] = NEG_BIG; lrow[j] = 0.0f; }

  const int nchunks = (q_base + 64) >> 5;      // causal: keys <= q_base+63
  stage(0, kvbuf[0]);

  for (int ci = 0; ci < nchunks; ++ci) {
    asm volatile("s_wait_asynccnt 0x0" ::: "memory");  // this wave's copies done
    __syncthreads();                                   // all waves' copies done;
                                                       // prev buffer fully consumed
    if (ci + 1 < nchunks) stage(ci + 1, kvbuf[(ci + 1) & 1]);

    const int kb = ci << 5;
    if (kb < qw_base + 16) {  // wave-uniform causal skip
      const float* kc = kvbuf[ci & 1];
      const float* vc = kc + CHUNKF;

      // ---- S = Q K^T : two 16x16 tiles, 4 WMMAs, B operands from LDS ----
      v16h b00, b01, b10, b11;
      {
        const float* kr0 = kc + (size_t)col * KSTRIDE + (hi << 4);
        const float* kr1 = kc + (size_t)(16 + col) * KSTRIDE + (hi << 4);
#pragma unroll
        for (int e = 0; e < 16; ++e) {   // contiguous -> ds_load_b128
          b00[e] = (_Float16)kr0[e];
          b01[e] = (_Float16)kr0[e + 32];
          b10[e] = (_Float16)kr1[e];
          b11[e] = (_Float16)kr1[e + 32];
        }
      }
      v8f s0 = {}, s1 = {};
      s0 = __builtin_amdgcn_wmma_f32_16x16x32_f16(false, aq0, false, b00, (short)0, s0, false, false);
      s0 = __builtin_amdgcn_wmma_f32_16x16x32_f16(false, aq1, false, b01, (short)0, s0, false, false);
      s1 = __builtin_amdgcn_wmma_f32_16x16x32_f16(false, aq0, false, b10, (short)0, s1, false, false);
      s1 = __builtin_amdgcn_wmma_f32_16x16x32_f16(false, aq1, false, b11, (short)0, s1, false, false);

      // ---- online softmax (fp32, DPP reductions) ----
      const bool need_mask = (kb + 31) > qw_base;
      float p0[8], p1[8], alpha[8];
#pragma unroll
      for (int j = 0; j < 8; ++j) {
        const int row = qw_base + j + (hi << 3);
        float x0 = s0[j], x1 = s1[j];
        if (need_mask) {
          if (kb + col > row)      x0 = NEG_BIG;
          if (kb + 16 + col > row) x1 = NEG_BIG;
        }
        float rm = red_max16(fmaxf(x0, x1));
        float mn = fmaxf(mrow[j], rm);
        float al = __expf(mrow[j] - mn);
        float e0 = __expf(x0 - mn);
        float e1 = __expf(x1 - mn);
        float rs = red_sum16(e0 + e1);
        lrow[j] = lrow[j] * al + rs;
        mrow[j] = mn;
        alpha[j] = al;
        p0[j] = e0;
        p1[j] = e1;
      }
#pragma unroll
      for (int j = 0; j < 8; ++j) {
        oacc[0][j] *= alpha[j]; oacc[1][j] *= alpha[j];
        oacc[2][j] *= alpha[j]; oacc[3][j] *= alpha[j];
      }

      // ---- transpose P (C layout -> A layout) through per-wave LDS ----
      float* lds = ptile[wib];
#pragma unroll
      for (int j = 0; j < 8; ++j) {
        const int row = j + (hi << 3);
        lds[row * PSTRIDE + col]      = p0[j];
        lds[row * PSTRIDE + 16 + col] = p1[j];
      }
      asm volatile("s_wait_dscnt 0x0" ::: "memory");  // DS in-order per wave
      v16h ap;
#pragma unroll
      for (int e = 0; e < 16; ++e) {
        int j  = e >> 1, lo = e & 1;
        int kk = (hi << 4) + e;
        ap[e] = (_Float16)lds[col * PSTRIDE + kk];
      }

      // ---- O += P * V : 4 dim-slices, 4 accumulating WMMAs (V from LDS) ----
#pragma unroll
      for (int t = 0; t < 4; ++t) {
        const int ns = t << 4;
        v16h bv;
#pragma unroll
        for (int e = 0; e < 16; ++e) {
          const int krow = (hi << 4) + e;  // key index within chunk
          bv[e] = (_Float16)vc[(size_t)krow * KSTRIDE + ns + col];
        }
        oacc[t] = __builtin_amdgcn_wmma_f32_16x16x32_f16(false, ap, false, bv,
                                                         (short)0, oacc[t], false, false);
      }
    }
  }

  // ---- normalize and write out ----
#pragma unroll
  for (int j = 0; j < 8; ++j) {
    const float inv = 1.0f / lrow[j];
    float* orow = Ob + (size_t)(qw_base + j + (hi << 3)) * HD;
    orow[col]      = oacc[0][j] * inv;
    orow[16 + col] = oacc[1][j] * inv;
    orow[32 + col] = oacc[2][j] * inv;
    orow[48 + col] = oacc[3][j] * inv;
  }
}

extern "C" void kernel_launch(void* const* d_in, const int* in_sizes, int n_in,
                              void* d_out, int out_size, void* d_ws, size_t ws_size,
                              hipStream_t stream) {
  (void)n_in; (void)out_size; (void)d_ws; (void)ws_size;
  const float* k = (const float*)d_in[0];  // setup_inputs order: k, q, v
  const float* q = (const float*)d_in[1];
  const float* v = (const float*)d_in[2];
  float* out = (float*)d_out;

  const int B = 4;
  const int T = in_sizes[0] / (B * HD);
  const float scale = 1.0f / sqrtf(512.0f);  // n_embed^-0.5

  const int blocks  = B * (T / 64);          // 4 wave32 per block, 64 q-rows
  flash_attn_causal_wmma_async<<<blocks, 128, 0, stream>>>(k, q, v, out, T, scale);
}
